// ChebConv_60825326846457
// MI455X (gfx1250) — matrix operand order repaired and checked by reading, not compiled, not run on hardware
//
#include <hip/hip_runtime.h>
#include <hip/hip_bf16.h>

typedef __attribute__((ext_vector_type(16))) _Float16 v16h;
typedef __attribute__((ext_vector_type(8)))  _Float16 v8h;
typedef __attribute__((ext_vector_type(8)))  float    v8f;
typedef __attribute__((ext_vector_type(4)))  float    v4f;

#define BB 8
#define NN 1024
#define CC 64
#define OO 64
#define MM 4
#define BN (BB*NN)
#define L2E 1.4426950408889634f

// ---------------- Kernel 1: attention projections ----------------
// One wave per (b,n) row. sR_i = Xr@Wr_i - Xi@Wi_i, etc. Bias folded into sRj/sIj.
__global__ void k_attn(const float* __restrict__ Xr, const float* __restrict__ Xi,
                       const float* __restrict__ awr, const float* __restrict__ awi,
                       const float* __restrict__ abr, const float* __restrict__ abi,
                       float* __restrict__ sRi, float* __restrict__ sIi,
                       float* __restrict__ sRj, float* __restrict__ sIj) {
    int lane = threadIdx.x & 31;
    int r = blockIdx.x * 8 + (threadIdx.x >> 5);      // b*N + n
    const float* xr = Xr + (size_t)r * CC;
    const float* xi = Xi + (size_t)r * CC;
    float xr0 = xr[lane], xr1 = xr[lane + 32];
    float xi0 = xi[lane], xi1 = xi[lane + 32];
    float wri0 = awr[lane], wri1 = awr[lane + 32];
    float wrj0 = awr[64 + lane], wrj1 = awr[96 + lane];
    float wii0 = awi[lane], wii1 = awi[lane + 32];
    float wij0 = awi[64 + lane], wij1 = awi[96 + lane];
    float pRi = xr0*wri0 + xr1*wri1 - xi0*wii0 - xi1*wii1;
    float pIi = xr0*wii0 + xr1*wii1 + xi0*wri0 + xi1*wri1;
    float pRj = xr0*wrj0 + xr1*wrj1 - xi0*wij0 - xi1*wij1;
    float pIj = xr0*wij0 + xr1*wij1 + xi0*wrj0 + xi1*wrj1;
    #pragma unroll
    for (int s = 16; s; s >>= 1) {
        pRi += __shfl_xor(pRi, s, 32);
        pIi += __shfl_xor(pIi, s, 32);
        pRj += __shfl_xor(pRj, s, 32);
        pIj += __shfl_xor(pIj, s, 32);
    }
    if (lane == 0) {
        sRi[r] = pRi; sIi[r] = pIi;
        sRj[r] = pRj + abr[0]; sIj[r] = pIj + abi[0];
    }
}

// ---------------- Kernel 2: softmax stats over axis i, per (b,j) ----------------
__global__ void k_softmax(const float* __restrict__ sRi, const float* __restrict__ sIi,
                          const float* __restrict__ sRj, const float* __restrict__ sIj,
                          const float* __restrict__ par, const float* __restrict__ pai,
                          float* __restrict__ mxl2, float* __restrict__ invden) {
    int lane = threadIdx.x & 31;
    int r = blockIdx.x * 8 + (threadIdx.x >> 5);      // b*N + j
    int b = r >> 10;
    float srj = sRj[r], sij = sIj[r];
    float a_r = par[0], a_i = pai[0];
    const float* pRi = sRi + (b << 10);
    const float* pIi = sIi + (b << 10);
    float mags[32];
    float vmax = -1e30f;
    #pragma unroll
    for (int t = 0; t < 32; ++t) {
        int i = lane + (t << 5);
        float sr = pRi[i] + srj;
        float si = pIi[i] + sij;
        float pr = sr >= 0.f ? sr : a_r * sr;
        float pi = si >= 0.f ? si : a_i * si;
        float mag = __builtin_amdgcn_sqrtf(pr*pr + pi*pi);
        mags[t] = mag;
        vmax = fmaxf(vmax, mag);
    }
    #pragma unroll
    for (int s = 16; s; s >>= 1) vmax = fmaxf(vmax, __shfl_xor(vmax, s, 32));
    float sum = 0.f;
    #pragma unroll
    for (int t = 0; t < 32; ++t)
        sum += __builtin_amdgcn_exp2f((mags[t] - vmax) * L2E);
    #pragma unroll
    for (int s = 16; s; s >>= 1) sum += __shfl_xor(sum, s, 32);
    if (lane == 0) { mxl2[r] = vmax * L2E; invden[r] = 1.0f / sum; }
}

// ---------------- Kernel W: transpose w -> f16 (m,o,c) for contiguous B fragments ----------------
__global__ void k_wprep(const float* __restrict__ wr, const float* __restrict__ wi,
                        _Float16* __restrict__ wrT, _Float16* __restrict__ wiT) {
    int idx = blockIdx.x * 256 + threadIdx.x;         // (m*64+o)*64+c, 16384 total
    int m = idx >> 12;
    int o = (idx >> 6) & 63;
    int c = idx & 63;
    int src = (m << 12) + (c << 6) + o;
    wrT[idx] = (_Float16)wr[src];
    wiT[idx] = (_Float16)wi[src];
}

// ---------------- Kernel 3: the 268 MB streaming pass over L ----------------
// One wave per (b,i). rowR[b,m,i] = sum_j Lr*ar - Li*ai ; rowI = sum_j Lr*ai + Li*ar.
// L is read exactly once and exceeds the 192 MB L2, so stream it with NT loads to
// keep the small reused attention arrays resident in cache.
__global__ void k_rows(const float* __restrict__ Lr_, const float* __restrict__ Li_,
                       const float* __restrict__ sRi, const float* __restrict__ sIi,
                       const float* __restrict__ sRj, const float* __restrict__ sIj,
                       const float* __restrict__ mxl2, const float* __restrict__ invden,
                       const float* __restrict__ par, const float* __restrict__ pai,
                       float* __restrict__ rowR, float* __restrict__ rowI) {
    int lane = threadIdx.x & 31;
    int r = blockIdx.x * 8 + (threadIdx.x >> 5);      // b*N + i
    int b = r >> 10;
    int i = r & 1023;
    float sric = sRi[r], siic = sIi[r];
    float a_r = par[0], a_i = pai[0];
    const v4f* pRj = (const v4f*)(sRj    + (b << 10));
    const v4f* pIj = (const v4f*)(sIj    + (b << 10));
    const v4f* pMx = (const v4f*)(mxl2   + (b << 10));
    const v4f* pDn = (const v4f*)(invden + (b << 10));
    const v4f* lr[MM]; const v4f* li[MM];
    #pragma unroll
    for (int m = 0; m < MM; ++m) {
        size_t off = (((size_t)(b*MM + m) << 10) + i) << 10;
        lr[m] = (const v4f*)(Lr_ + off);
        li[m] = (const v4f*)(Li_ + off);
    }
    float accR[MM] = {0,0,0,0}, accI[MM] = {0,0,0,0};
    for (int jt = 0; jt < 8; ++jt) {
        int q = lane + (jt << 5);                     // float4 index: j = 4q..4q+3
        v4f srj = pRj[q], sij = pIj[q], mx = pMx[q], idn = pDn[q];
        float arv[4], aiv[4];
        #pragma unroll
        for (int e = 0; e < 4; ++e) {
            float sr = sric + srj[e];
            float si = siic + sij[e];
            float pr = sr >= 0.f ? sr : a_r * sr;
            float pi = si >= 0.f ? si : a_i * si;
            float mag = __builtin_amdgcn_sqrtf(pr*pr + pi*pi);
            float sm  = __builtin_amdgcn_exp2f(mag * L2E - mx[e]) * idn[e];
            float sc  = sm * __builtin_amdgcn_rcpf(mag + 1e-12f);
            arv[e] = sc * pr;
            aiv[e] = sc * pi;
        }
        #pragma unroll
        for (int m = 0; m < MM; ++m) {
            v4f L4 = __builtin_nontemporal_load(lr[m] + q);
            v4f I4 = __builtin_nontemporal_load(li[m] + q);
            #pragma unroll
            for (int e = 0; e < 4; ++e) {
                accR[m] += L4[e]*arv[e] - I4[e]*aiv[e];
                accI[m] += L4[e]*aiv[e] + I4[e]*arv[e];
            }
        }
    }
    #pragma unroll
    for (int s = 16; s; s >>= 1) {
        #pragma unroll
        for (int m = 0; m < MM; ++m) {
            accR[m] += __shfl_xor(accR[m], s, 32);
            accI[m] += __shfl_xor(accI[m], s, 32);
        }
    }
    if (lane == 0) {
        #pragma unroll
        for (int m = 0; m < MM; ++m) {
            rowR[((b*MM + m) << 10) + i] = accR[m];
            rowI[((b*MM + m) << 10) + i] = accI[m];
        }
    }
}

// ---------------- Kernel 4: WMMA final projection ----------------
// One wave per 16x16 output tile (b, jtile, otile). P = Xr@wr[m], Q = Xi@wi[m] via
// v_wmma_f32_16x16x32_f16 (two K=32 steps), then f32 combine with rowR/rowI.
__global__ void k_out(const float* __restrict__ Xr, const float* __restrict__ Xi,
                      const _Float16* __restrict__ wrT, const _Float16* __restrict__ wiT,
                      const float* __restrict__ rowR, const float* __restrict__ rowI,
                      float* __restrict__ outRe, float* __restrict__ outIm) {
    int lane = threadIdx.x & 31;
    int w = (blockIdx.x * blockDim.x + threadIdx.x) >> 5;   // 0..2047
    int b = w >> 8;
    int rem = w & 255;
    int jt = rem >> 2, ot = rem & 3;
    int j0 = jt << 4, o0 = ot << 4;
    int halfSel = lane >> 4;        // lanes 16-31 hold the other K half
    int lr16 = lane & 15;
    int arow = j0 + lr16;           // A row (M) held by this lane
    int bcol = o0 + lr16;           // B column (N) held by this lane
    v8f accRe = {}, accIm = {};
    #pragma unroll
    for (int m = 0; m < MM; ++m) {
        v8f P = {}, Q = {};
        #pragma unroll
        for (int ks = 0; ks < 2; ++ks) {
            int k0 = ks << 5;
            // A fragment (16-bit A 16x32 layout): lanes<16: K = k0+{0..7, 16..23};
            // lanes>=16: K = k0+{8..15, 24..31}
            const float* ar_p = Xr + (((size_t)(b << 10) + arow) << 6) + k0 + (halfSel << 3);
            const float* ai_p = Xi + (((size_t)(b << 10) + arow) << 6) + k0 + (halfSel << 3);
            v16h Ar, Ai;
            #pragma unroll
            for (int t = 0; t < 8; ++t) {
                Ar[t]     = (_Float16)ar_p[t];
                Ar[t + 8] = (_Float16)ar_p[t + 16];
                Ai[t]     = (_Float16)ai_p[t];
                Ai[t + 8] = (_Float16)ai_p[t + 16];
            }
            // B fragment: lane holds column bcol, K = k0 + halfSel*8 + {0..7, 16..23},
            // from the (m,o,c) transposed f16 copy.
            const v8h* br_p = (const v8h*)(wrT + (((m << 6) + bcol) << 6) + k0 + (halfSel << 3));
            const v8h* bi_p = (const v8h*)(wiT + (((m << 6) + bcol) << 6) + k0 + (halfSel << 3));
            v8h b0 = br_p[0], b1 = br_p[2];
            v8h c0 = bi_p[0], c1 = bi_p[2];
            v16h Br, Bi;
            #pragma unroll
            for (int t = 0; t < 8; ++t) { Br[t] = b0[t]; Br[t + 8] = b1[t]; }
            #pragma unroll
            for (int t = 0; t < 8; ++t) { Bi[t] = c0[t]; Bi[t + 8] = c1[t]; }
            P = __builtin_amdgcn_wmma_f32_16x16x32_f16(false, Ar, false, Br, (short)0, P, false, false);
            Q = __builtin_amdgcn_wmma_f32_16x16x32_f16(false, Ai, false, Bi, (short)0, Q, false, false);
        }
        // C/D layout: VGPR t -> M = j0 + t + 8*halfSel, N = o0 + (lane&15)
        const float* rRp = rowR + (((b << 2) + m) << 10);
        const float* rIp = rowI + (((b << 2) + m) << 10);
        #pragma unroll
        for (int t = 0; t < 8; ++t) {
            int j = j0 + t + (halfSel << 3);
            float rR = rRp[j], rI = rIp[j];
            accRe[t] += rR * P[t] - rI * Q[t];
            accIm[t] += rI * P[t] + rR * Q[t];
        }
    }
    #pragma unroll
    for (int t = 0; t < 8; ++t) {
        int j = j0 + t + (halfSel << 3);
        size_t off = (((size_t)(b << 10) + j) << 6) + o0 + lr16;
        __builtin_nontemporal_store(accRe[t], outRe + off);
        __builtin_nontemporal_store(accIm[t], outIm + off);
    }
}

extern "C" void kernel_launch(void* const* d_in, const int* in_sizes, int n_in,
                              void* d_out, int out_size, void* d_ws, size_t ws_size,
                              hipStream_t stream) {
    const float* X_real = (const float*)d_in[0];
    const float* X_imag = (const float*)d_in[1];
    const float* L_real = (const float*)d_in[2];
    const float* L_imag = (const float*)d_in[3];
    const float* w_real = (const float*)d_in[4];
    const float* w_imag = (const float*)d_in[5];
    const float* aw_r   = (const float*)d_in[6];
    const float* aw_i   = (const float*)d_in[7];
    const float* ab_r   = (const float*)d_in[8];
    const float* ab_i   = (const float*)d_in[9];
    const float* pa_r   = (const float*)d_in[10];
    const float* pa_i   = (const float*)d_in[11];

    float* ws = (float*)d_ws;
    float* sRi    = ws;                 // 8192
    float* sIi    = ws + 1 * BN;
    float* sRj    = ws + 2 * BN;
    float* sIj    = ws + 3 * BN;
    float* mxl2   = ws + 4 * BN;
    float* invden = ws + 5 * BN;
    float* rowR   = ws + 6 * BN;        // 32768
    float* rowI   = rowR + MM * BN;     // 32768
    _Float16* wrT = (_Float16*)(rowI + MM * BN);   // 16384 halves
    _Float16* wiT = wrT + MM * CC * OO;            // 16384 halves

    float* outRe = (float*)d_out;
    float* outIm = outRe + (size_t)BB * NN * OO;

    k_attn<<<BN / 8, 256, 0, stream>>>(X_real, X_imag, aw_r, aw_i, ab_r, ab_i,
                                       sRi, sIi, sRj, sIj);
    k_softmax<<<BN / 8, 256, 0, stream>>>(sRi, sIi, sRj, sIj, pa_r, pa_i, mxl2, invden);
    k_wprep<<<(MM * CC * OO) / 256, 256, 0, stream>>>(w_real, w_imag, wrT, wiT);
    k_rows<<<BN / 8, 256, 0, stream>>>(L_real, L_imag, sRi, sIi, sRj, sIj,
                                       mxl2, invden, pa_r, pa_i, rowR, rowI);
    k_out<<<256, 256, 0, stream>>>(X_real, X_imag, wrT, wiT, rowR, rowI, outRe, outIm);
}
